// CrossNetMix_42941083025844
// MI455X (gfx1250) — matrix operand order repaired and checked by reading, not compiled, not run on hardware
//
#include <hip/hip_runtime.h>
#include <hip/hip_bf16.h>

// ---------------------------------------------------------------------------
// CrossNetMix (DCN-v2 mixture-of-low-rank-experts) on MI455X / gfx1250.
//
// Math per layer (row-local):  g = softmax(xl Gw^T);  t1 = tanh(xl V^T);
// t2 = tanh(C t1);  u = g (x) t2 (gate folded in since sum_e g = 1);
// xl += x0 .* (U_cat u + bias).
//
// Engine choice (roofline): 54 GFLOP vs 128 MB HBM -> AI ~420 FLOP/B, compute
// bound. Use v_wmma_f32_16x16x32_f16 (f16 A/B, fp32 accumulate) for both big
// GEMMs + the block-diag C GEMM; xl/residual/tanh/softmax stay fp32 in LDS.
// Weights (3.1 MB as f16 fragments) are pre-laid-out in d_ws and live in L2.
// x tile is staged into LDS via the Tensor Data Mover (tensor_load_to_lds +
// s_wait_tensorcnt). tanh is a branch-free v_exp_f32 + v_rcp_f32 sequence so
// waves stay convergent between WMMA stages.
// ---------------------------------------------------------------------------

typedef __attribute__((ext_vector_type(16))) _Float16 v16h;
typedef __attribute__((ext_vector_type(8)))  _Float16 v8h;
typedef __attribute__((ext_vector_type(8)))  float    v8f;
typedef __attribute__((ext_vector_type(4)))  float    v4f;
typedef __attribute__((ext_vector_type(4)))  unsigned int v4u;
typedef __attribute__((ext_vector_type(8)))  int      v8i;
typedef __attribute__((ext_vector_type(4)))  int      v4i;

#define LAYERS  3
#define EXPERTS 4
#define FDIM    1024
#define BROWS   16384
#define BM      32          // rows per workgroup

// LDS strides (floats / halves) chosen so 16 consecutive rows hit distinct banks
#define XLS 1028            // fp32 xl row stride: 1028 % 64 == 4 -> conflict-free
#define T1S 264             // f16 row stride: 528B -> 4-bank row skew

// LDS byte layout (dynamic shared, base offset 0)
#define XL_OFF 0
#define X0_OFF (XL_OFF + BM * XLS * 4)
#define T1_OFF (X0_OFF + BM * FDIM * 4)
#define UH_OFF (T1_OFF + BM * T1S * 2)
#define G_OFF  (UH_OFF + BM * T1S * 2)
#define LDS_BYTES (G_OFF + BM * EXPERTS * 4)   // ~290 KB of 320 KB WGP LDS

// d_ws layout (f16 element offsets). Each B-fragment block = 32 lanes x 16 halves.
#define VT_OFF 0ull
#define VT_HALVES ((size_t)LAYERS * 16 * 32 * 512)   // [l][ntile16][kstep32]
#define UT_OFF (VT_OFF + VT_HALVES)
#define UT_HALVES ((size_t)LAYERS * 64 * 8 * 512)    // [l][ntile64][kstep8]
#define CT_OFF (UT_OFF + UT_HALVES)
#define CT_HALVES ((size_t)LAYERS * 4 * 4 * 2 * 512) // [l][e][ntile4][kstep2]
#define W_TOTAL (CT_OFF + CT_HALVES)                 // 1,622,016 halves (~3.1 MB)

// Branch-free tanh: clamp to +-9 (tanh saturates to 1 ulp well before),
// then (e^2x - 1) * rcp(e^2x + 1). v_exp_f32 + v_rcp_f32, no EXEC divergence.
__device__ __forceinline__ float fast_tanh(float v) {
    float c = fminf(fmaxf(v, -9.f), 9.f);
    float t = __expf(2.f * c);
    return (t - 1.f) * __builtin_amdgcn_rcpf(t + 1.f);
}

// ---------------------------------------------------------------------------
// Weight prep: f32 -> f16 B-fragments in WMMA hardware order.
// B (32x16, K x N) layout: N = lane%16, K = 8*(lane/16) + (slot&7) + 16*(slot>>3)
// (VGPR j holds K=8h+2j,8h+2j+1 for j<4, 8+8h+2j,9+8h+2j for j>=4; h=lane/16).
// ---------------------------------------------------------------------------
__global__ void crossnet_prep(const float* __restrict__ U, const float* __restrict__ V,
                              const float* __restrict__ C, _Float16* __restrict__ W) {
    size_t t = (size_t)blockIdx.x * 256 + threadIdx.x;
    if (t >= W_TOTAL) return;
    int s    = (int)(t & 15);          // f16 slot within lane
    int lane = (int)((t >> 4) & 31);
    size_t blk = t >> 9;               // fragment block index
    int kloc = ((lane >> 4) << 3) + (s & 7) + ((s >> 3) << 4); // K within 32-row block (table order)
    int n    = lane & 15;              // N within 16-col block
    float v;
    if (t < UT_OFF) {                          // Vt: [F=1024] x [E*R=256]
        int k  = (int)(blk & 31);
        int nt = (int)((blk >> 5) & 15);
        int l  = (int)(blk >> 9);
        int kk = k * 32 + kloc;                // f
        int nn = nt * 16 + n;                  // e*64 + r
        int e = nn >> 6, r = nn & 63;
        v = V[(((size_t)l * EXPERTS + e) * 64 + r) * FDIM + kk];
    } else if (t < CT_OFF) {                   // Ut: [E*R=256] x [F=1024]
        size_t b2 = blk - (UT_OFF >> 9);
        int k  = (int)(b2 & 7);
        int nt = (int)((b2 >> 3) & 63);
        int l  = (int)(b2 >> 9);
        int kk = k * 32 + kloc;                // e*64 + s
        int nn = nt * 16 + n;                  // f
        int e = kk >> 6, ss = kk & 63;
        v = U[(((size_t)l * EXPERTS + e) * FDIM + nn) * 64 + ss];
    } else {                                   // Ct: per expert [R=64] x [R=64]
        size_t b2 = blk - (CT_OFF >> 9);
        int k  = (int)(b2 & 1);
        int nt = (int)((b2 >> 1) & 3);
        int e  = (int)((b2 >> 3) & 3);
        int l  = (int)(b2 >> 5);
        int kk = k * 32 + kloc;                // r
        int nn = nt * 16 + n;                  // s
        v = C[(((size_t)l * EXPERTS + e) * 64 + nn) * 64 + kk];
    }
    W[t] = (_Float16)v;
}

// ---------------------------------------------------------------------------
// Main kernel: 256 threads = 8 wave32; wave = (strip: rows 16, quad: N chunk).
// ---------------------------------------------------------------------------
__global__ __launch_bounds__(256, 1)
void crossnet_main(const float* __restrict__ x, const float* __restrict__ bias,
                   const float* __restrict__ gate_w,
                   const _Float16* __restrict__ Vt, const _Float16* __restrict__ Ut,
                   const _Float16* __restrict__ Ct, float* __restrict__ out) {
    extern __shared__ char smem[];
    float*    xl  = (float*)(smem + XL_OFF);
    float*    x0  = (float*)(smem + X0_OFF);
    _Float16* t1h = (_Float16*)(smem + T1_OFF);
    _Float16* uh  = (_Float16*)(smem + UH_OFF);
    float*    gl  = (float*)(smem + G_OFF);

    const int tid   = threadIdx.x;
    const int lane  = tid & 31;
    const int wave  = tid >> 5;
    const int strip = wave >> 2;        // 0..1 : rows [strip*16, +16)
    const int quad  = wave & 3;         // 0..3 : N partition
    const int mrow  = lane & 15;
    const int half  = lane >> 4;
    const size_t row0 = (size_t)blockIdx.x * BM;

    // ---- TDM: stage x tile [BM x 1024] f32 into LDS (x0), wave 0 issues ----
    if (wave == 0) {
        unsigned long long ga = (unsigned long long)(const void*)(x + row0 * FDIM);
        v4u g0 = { 1u,                                      // count = 1 (valid D#)
                   (unsigned)X0_OFF,                        // lds_addr
                   (unsigned)(ga & 0xFFFFFFFFull),          // global_addr[31:0]
                   (unsigned)((ga >> 32) & 0x1FFFFFFull) | (2u << 30) }; // [56:32] | type=2
        v8i g1 = { (int)(2u << 16),                         // data_size = 4B
                   (int)((FDIM & 0xFFFF) << 16),            // tensor_dim0 lo16
                   (int)((FDIM >> 16) | ((BROWS & 0xFFFF) << 16)), // dim0 hi | dim1 lo
                   (int)((BROWS >> 16) | (FDIM << 16)),     // dim1 hi | tile_dim0 = 1024
                   BM,                                      // tile_dim1 = 32, tile_dim2 = 0
                   FDIM,                                    // tensor_dim0_stride lo32
                   0, 0 };
        v4i g2 = { 0, 0, 0, 0 };
        v4i g3 = { 0, 0, 0, 0 };
        v8i g4 = { 0, 0, 0, 0, 0, 0, 0, 0 };                // extra group (6-arg form)
        __builtin_amdgcn_tensor_load_to_lds(g0, g1, g2, g3, g4, 0);
        __builtin_amdgcn_s_wait_tensorcnt(0);
    }
    __syncthreads();

    // xl = x0 (re-strided fp32 working copy)
    for (int idx = tid; idx < BM * FDIM; idx += 256) {
        int r = idx >> 10, c = idx & (FDIM - 1);
        xl[r * XLS + c] = x0[r * FDIM + c];
    }
    __syncthreads();

    for (int l = 0; l < LAYERS; ++l) {
        // ---- stage 1: gate logits + softmax (fp32) ----
        if (tid < BM * EXPERTS) {
            int r = tid >> 2, e = tid & 3;
            const float* xr = xl + r * XLS;
            const float* gw = gate_w + e * FDIM;
            float s = 0.f;
#pragma unroll 4
            for (int c = 0; c < FDIM; c += 4) {
                v4f a = *(const v4f*)(xr + c);
                v4f b = *(const v4f*)(gw + c);
                s += a[0] * b[0] + a[1] * b[1] + a[2] * b[2] + a[3] * b[3];
            }
            gl[tid] = s;
        }
        __syncthreads();
        if (tid < BM) {
            float a0 = gl[tid * 4 + 0], a1 = gl[tid * 4 + 1];
            float a2 = gl[tid * 4 + 2], a3 = gl[tid * 4 + 3];
            float m = fmaxf(fmaxf(a0, a1), fmaxf(a2, a3));
            float e0 = __expf(a0 - m), e1 = __expf(a1 - m);
            float e2 = __expf(a2 - m), e3 = __expf(a3 - m);
            float inv = __builtin_amdgcn_rcpf(e0 + e1 + e2 + e3);
            gl[tid * 4 + 0] = e0 * inv; gl[tid * 4 + 1] = e1 * inv;
            gl[tid * 4 + 2] = e2 * inv; gl[tid * 4 + 3] = e3 * inv;
        }

        // ---- stage 2: t1 = tanh(xl @ Vt)  [BMx1024]x[1024x256] ----
        {
            const float* xr = xl + (strip * 16 + mrow) * XLS;
            v8f acc[4] = {};
            for (int k = 0; k < 32; ++k) {
                int c0 = k * 32 + half * 8;    // A-frag K runs {8h..8h+7, 16+8h..23+8h}
                v16h a;
#pragma unroll
                for (int s = 0; s < 8; ++s) a[s]     = (_Float16)xr[c0 + s];
#pragma unroll
                for (int s = 0; s < 8; ++s) a[s + 8] = (_Float16)xr[c0 + 16 + s];
#pragma unroll
                for (int nt = 0; nt < 4; ++nt) {
                    const v16h b = *(const v16h*)(Vt +
                        ((((size_t)l * 16 + quad * 4 + nt) * 32 + k) << 9) + lane * 16);
                    acc[nt] = __builtin_amdgcn_wmma_f32_16x16x32_f16(
                        false, a, false, b, (short)0, acc[nt], false, false);
                }
            }
#pragma unroll
            for (int nt = 0; nt < 4; ++nt)
#pragma unroll
                for (int j = 0; j < 8; ++j) {
                    int r = strip * 16 + j + 8 * half;
                    int c = quad * 64 + nt * 16 + (lane & 15);
                    t1h[r * T1S + c] = (_Float16)fast_tanh(acc[nt][j]);
                }
        }
        __syncthreads();

        // ---- stage 3: u = g (x) tanh(C @ t1), block-diag per expert ----
        {
            const int e = quad;
            v8f acc[4] = {};
#pragma unroll
            for (int k = 0; k < 2; ++k) {
                const _Float16* ar = t1h + (strip * 16 + mrow) * T1S + e * 64 + k * 32 + half * 8;
                v8h lo = *(const v8h*)ar;
                v8h hi = *(const v8h*)(ar + 16);
                v16h a = __builtin_shufflevector(lo, hi, 0,1,2,3,4,5,6,7,8,9,10,11,12,13,14,15);
#pragma unroll
                for (int nt = 0; nt < 4; ++nt) {
                    const v16h b = *(const v16h*)(Ct +
                        (((((size_t)l * 4 + e) * 4 + nt) * 2 + k) << 9) + lane * 16);
                    acc[nt] = __builtin_amdgcn_wmma_f32_16x16x32_f16(
                        false, a, false, b, (short)0, acc[nt], false, false);
                }
            }
#pragma unroll
            for (int nt = 0; nt < 4; ++nt)
#pragma unroll
                for (int j = 0; j < 8; ++j) {
                    int r = strip * 16 + j + 8 * half;
                    int c = e * 64 + nt * 16 + (lane & 15);
                    float gv = gl[r * 4 + e];
                    uh[r * T1S + c] = (_Float16)(gv * fast_tanh(acc[nt][j]));
                }
        }
        __syncthreads();

        // ---- stage 4: xl += x0 .* (u @ Ut + bias)  [BMx256]x[256x1024] ----
        {
            v16h af[8];                       // preload all A fragments (K=256)
#pragma unroll
            for (int k = 0; k < 8; ++k) {
                const _Float16* ar = uh + (strip * 16 + mrow) * T1S + k * 32 + half * 8;
                v8h lo = *(const v8h*)ar;
                v8h hi = *(const v8h*)(ar + 16);
                af[k] = __builtin_shufflevector(lo, hi, 0,1,2,3,4,5,6,7,8,9,10,11,12,13,14,15);
            }
            for (int nt = 0; nt < 16; ++nt) {
                v8f acc = {};
#pragma unroll
                for (int k = 0; k < 8; ++k) {
                    const v16h b = *(const v16h*)(Ut +
                        ((((size_t)l * 64 + quad * 16 + nt) * 8 + k) << 9) + lane * 16);
                    acc = __builtin_amdgcn_wmma_f32_16x16x32_f16(
                        false, af[k], false, b, (short)0, acc, false, false);
                }
                int c = quad * 256 + nt * 16 + (lane & 15);
                float bv = bias[l * FDIM + c];
#pragma unroll
                for (int j = 0; j < 8; ++j) {
                    int r = strip * 16 + j + 8 * half;
                    xl[r * XLS + c] += x0[r * FDIM + c] * (acc[j] + bv);
                }
            }
        }
        __syncthreads();
    }

    // ---- write result (streaming, non-temporal) ----
    for (int idx = tid; idx < BM * FDIM; idx += 256) {
        int r = idx >> 10, c = idx & (FDIM - 1);
        __builtin_nontemporal_store(xl[r * XLS + c], out + (row0 + r) * FDIM + c);
    }
}

extern "C" void kernel_launch(void* const* d_in, const int* in_sizes, int n_in,
                              void* d_out, int out_size, void* d_ws, size_t ws_size,
                              hipStream_t stream) {
    (void)in_sizes; (void)n_in; (void)out_size; (void)ws_size;
    const float* x    = (const float*)d_in[0];
    const float* U    = (const float*)d_in[1];
    const float* V    = (const float*)d_in[2];
    const float* C    = (const float*)d_in[3];
    const float* bias = (const float*)d_in[4];
    const float* gw   = (const float*)d_in[5];
    _Float16* W = (_Float16*)d_ws;   // needs ~3.2 MB of workspace

    // weight f32 -> f16 fragment layout (runs every call; deterministic)
    int prep_blocks = (int)((W_TOTAL + 255) / 256);
    crossnet_prep<<<prep_blocks, 256, 0, stream>>>(U, V, C, W);

    (void)hipFuncSetAttribute((const void*)crossnet_main,
                              hipFuncAttributeMaxDynamicSharedMemorySize, LDS_BYTES);
    crossnet_main<<<BROWS / BM, 256, LDS_BYTES, stream>>>(
        x, bias, gw, W + VT_OFF, W + UT_OFF, W + CT_OFF, (float*)d_out);
}
